// Codebook_6047313952813
// MI455X (gfx1250) — matrix operand order repaired and checked by reading, not compiled, not run on hardware
//
#include <hip/hip_runtime.h>

// ---------------------------------------------------------------------------
// VQ-VAE codebook lookup for MI455X (gfx1250, wave32, WMMA).
// N = 16384 points, K = 8192 codes, D = 256.
// dist-argmin via bf16 hi/lo split GEMM (3x v_wmma_f32_16x16x32_bf16),
// async global->LDS double-buffered staging, coalesced transpose pre/post,
// exact fp32 loss.
// ---------------------------------------------------------------------------

typedef __attribute__((ext_vector_type(16))) __bf16 v16bf;
typedef __attribute__((ext_vector_type(8)))  float  v8f;

#define NPTS   16384
#define NCODE  8192
#define DDIM   256
#define APITCH 40   // bf16 units per LDS row: 32 data + 8 pad (80B, conflict-free)

// ---- async global->LDS staging path (device only, inline asm) --------------
// Per ISA: VDST = LDS byte offset (32-bit VGPR), VADDR = 64-bit global address
// pair, GV mode => "off" saddr.
#if defined(__HIP_DEVICE_COMPILE__) && defined(__gfx1250__)
  #define VQ_ASYNC 1
  #define ASYNC_B128(gp, lp)                                                   \
      asm volatile("global_load_async_to_lds_b128 %0, %1, off"                 \
                   :: "v"((unsigned)(size_t)(lp)),                             \
                      "v"((unsigned long long)(size_t)(gp)) : "memory")
#else
  #define VQ_ASYNC 0
#endif

#if defined(__HIP_DEVICE_COMPILE__) && __has_builtin(__builtin_amdgcn_s_wait_asynccnt)
  #define WAIT_ASYNC0() __builtin_amdgcn_s_wait_asynccnt(0)
#elif defined(__HIP_DEVICE_COMPILE__) && defined(__gfx1250__)
  #define WAIT_ASYNC0() asm volatile("s_wait_asynccnt 0x0" ::: "memory")
#else
  #define WAIT_ASYNC0()
#endif

union FragBF { uint4 u[2]; v16bf v; };

struct __align__(16) SMem {
    __bf16 A[2][2][128 * APITCH];   // [buffer][hi/lo]
    __bf16 B[2][2][128 * APITCH];   // [buffer][hi/lo]
    float  en[128];
    float  wv[8][64];
    int    wi[8][64];
};

// --------------------------- init best array --------------------------------
__global__ void vq_init_best(unsigned long long* __restrict__ best) {
    int t = blockIdx.x * 256 + threadIdx.x;
    if (t < NPTS) best[t] = 0xFFFFFFFFFFFFFFFFull;
}

// ------- transpose x [B,C,H,W] -> [N,D] + bf16 hi/lo split (coalesced) ------
// grid = 512 blocks, one per (b,h); block stages a 256(c) x 32(w) fp32 tile.
__global__ __launch_bounds__(256)
void vq_prep_x(const float* __restrict__ x,
               __bf16* __restrict__ Xhi, __bf16* __restrict__ Xlo) {
    __shared__ float tile[256 * 33];           // [c][w], pitch 33
    const int bh  = blockIdx.x;                // b*32 + h
    const int tid = threadIdx.x;

    // load: lanes vary w -> fully coalesced 128B per warp
    {
        const int w  = tid & 31;
        const int c0 = tid >> 5;               // 0..7
        #pragma unroll
        for (int j = 0; j < 32; ++j) {
            const int c = c0 + j * 8;
            tile[c * 33 + w] = x[(((size_t)bh >> 5) * DDIM + c) * 1024 +
                                 (((size_t)bh & 31) << 5) + w];
        }
    }
    __syncthreads();

    // store: each thread emits a contiguous 32-element (64B) bf16 run along d
    const int nloc = tid >> 3;                 // w  (0..31)
    const int a    = tid & 7;                  // c block (0..7)
    const int n    = (bh << 5) + nloc;         // b*1024 + h*32 + w
    __align__(16) __bf16 hi[32];
    __align__(16) __bf16 lo[32];
    #pragma unroll
    for (int j = 0; j < 32; ++j) {
        const float v = tile[(a * 32 + j) * 33 + nloc];
        const __bf16 h = (__bf16)v;
        hi[j] = h;
        lo[j] = (__bf16)(v - (float)h);
    }
    uint4* dhi = (uint4*)&Xhi[(size_t)n * DDIM + a * 32];
    uint4* dlo = (uint4*)&Xlo[(size_t)n * DDIM + a * 32];
    #pragma unroll
    for (int q = 0; q < 4; ++q) {
        dhi[q] = ((const uint4*)hi)[q];
        dlo[q] = ((const uint4*)lo)[q];
    }
}

// ---------------- emb bf16 hi/lo split + ||e||^2 ----------------------------
__global__ __launch_bounds__(256)
void vq_prep_emb(const float* __restrict__ emb,
                 __bf16* __restrict__ Ehi, __bf16* __restrict__ Elo,
                 float* __restrict__ enorm) {
    const int k = blockIdx.x;
    const int d = threadIdx.x;
    const float v = emb[(size_t)k * DDIM + d];
    const __bf16 h = (__bf16)v;
    const __bf16 l = (__bf16)(v - (float)h);
    Ehi[(size_t)k * DDIM + d] = h;
    Elo[(size_t)k * DDIM + d] = l;
    __shared__ float red[256];
    red[d] = v * v;
    __syncthreads();
    for (int s = 128; s > 0; s >>= 1) {
        if (d < s) red[d] += red[d + s];
        __syncthreads();
    }
    if (d == 0) enorm[k] = red[0];
}

// ---------------- staging helpers ------------------------------------------
__device__ __forceinline__
void vq_stage(SMem& sm, int buf,
              const __bf16* __restrict__ Xhi, const __bf16* __restrict__ Xlo,
              const __bf16* __restrict__ Ehi, const __bf16* __restrict__ Elo,
              int m0, int c0, int kk, int tid) {
    for (int e2 = tid; e2 < 512; e2 += 256) {
        const int row = e2 >> 2, q = e2 & 3;
        const size_t ga = (size_t)(m0 + row) * DDIM + kk + q * 8;
        const size_t gb = (size_t)(c0 + row) * DDIM + kk + q * 8;
        const int lo = row * APITCH + q * 8;
#if VQ_ASYNC
        ASYNC_B128(Xhi + ga, &sm.A[buf][0][lo]);
        ASYNC_B128(Xlo + ga, &sm.A[buf][1][lo]);
        ASYNC_B128(Ehi + gb, &sm.B[buf][0][lo]);
        ASYNC_B128(Elo + gb, &sm.B[buf][1][lo]);
#else
        *(uint4*)&sm.A[buf][0][lo] = *(const uint4*)(Xhi + ga);
        *(uint4*)&sm.A[buf][1][lo] = *(const uint4*)(Xlo + ga);
        *(uint4*)&sm.B[buf][0][lo] = *(const uint4*)(Ehi + gb);
        *(uint4*)&sm.B[buf][1][lo] = *(const uint4*)(Elo + gb);
#endif
    }
}

__device__ __forceinline__
void vq_compute(const SMem& sm, int buf, v8f acc[4][2],
                int wm, int wn, int half, int lrow) {
    FragBF ah[4], al[4], bh[2], bl[2];
    // A fragments: lane holds M=lrow, K runs [half*8..+7] and [16+half*8..+7]
    #pragma unroll
    for (int mi = 0; mi < 4; ++mi) {
        const int r = wm * 64 + mi * 16 + lrow;
        const int o = r * APITCH + half * 8;
        ah[mi].u[0] = *(const uint4*)&sm.A[buf][0][o];
        ah[mi].u[1] = *(const uint4*)&sm.A[buf][0][o + 16];
        al[mi].u[0] = *(const uint4*)&sm.A[buf][1][o];
        al[mi].u[1] = *(const uint4*)&sm.A[buf][1][o + 16];
    }
    // B fragments: lane holds code N=lrow, K runs [half*8..+7] and [16+half*8..+7]
    #pragma unroll
    for (int ni = 0; ni < 2; ++ni) {
        const int r = wn * 32 + ni * 16 + lrow;
        const int o = r * APITCH + half * 8;
        bh[ni].u[0] = *(const uint4*)&sm.B[buf][0][o];
        bh[ni].u[1] = *(const uint4*)&sm.B[buf][0][o + 16];
        bl[ni].u[0] = *(const uint4*)&sm.B[buf][1][o];
        bl[ni].u[1] = *(const uint4*)&sm.B[buf][1][o + 16];
    }
    // split-precision accumulation: hi*hi + hi*lo + lo*hi
    #pragma unroll
    for (int mi = 0; mi < 4; ++mi)
        #pragma unroll
        for (int ni = 0; ni < 2; ++ni) {
            acc[mi][ni] = __builtin_amdgcn_wmma_f32_16x16x32_bf16(
                false, ah[mi].v, false, bh[ni].v, (short)0, acc[mi][ni], false, false);
            acc[mi][ni] = __builtin_amdgcn_wmma_f32_16x16x32_bf16(
                false, ah[mi].v, false, bl[ni].v, (short)0, acc[mi][ni], false, false);
            acc[mi][ni] = __builtin_amdgcn_wmma_f32_16x16x32_bf16(
                false, al[mi].v, false, bh[ni].v, (short)0, acc[mi][ni], false, false);
        }
}

// ---------------- fused split-bf16 GEMM + argmin ----------------------------
// grid = (NPTS/128, NCODE/1024), block = 256 (8 waves: 2 M-groups x 4 N-groups)
__global__ __launch_bounds__(256)
void vq_gemm_argmin(const __bf16* __restrict__ Xhi, const __bf16* __restrict__ Xlo,
                    const __bf16* __restrict__ Ehi, const __bf16* __restrict__ Elo,
                    const float* __restrict__ enorm,
                    unsigned long long* __restrict__ best) {
    __shared__ SMem sm;
    const int tid  = threadIdx.x;
    const int lane = tid & 31;
    const int w    = tid >> 5;
    const int wm   = w & 1;          // M group: rows wm*64 .. +63
    const int wn   = w >> 1;         // N group: codes wn*32 .. +31
    const int half = lane >> 4;
    const int lrow = lane & 15;
    const int m0   = blockIdx.x * 128;
    const int n0   = blockIdx.y * 1024;

    for (int e = tid; e < 512; e += 256) {
        ((float*)sm.wv)[e] = 3.0e38f;
        ((int*)sm.wi)[e]   = 0x7fffffff;
    }

    for (int nt = 0; nt < 8; ++nt) {
        const int c0 = n0 + nt * 128;

        v8f acc[4][2];
        const v8f vzero = {0.f, 0.f, 0.f, 0.f, 0.f, 0.f, 0.f, 0.f};
        #pragma unroll
        for (int mi = 0; mi < 4; ++mi)
            #pragma unroll
            for (int ni = 0; ni < 2; ++ni) acc[mi][ni] = vzero;

        __syncthreads();   // previous nt: epilogue done, buffers free

#if VQ_ASYNC
        vq_stage(sm, 0, Xhi, Xlo, Ehi, Elo, m0, c0, 0, tid);
        if (tid < 128) sm.en[tid] = enorm[c0 + tid];
        int cur = 0;
        for (int kk = 0; kk < DDIM; kk += 32) {
            WAIT_ASYNC0();          // this wave's batch for buf[cur] landed
            __syncthreads();        // everyone's batch landed; prev compute done
            if (kk + 32 < DDIM)
                vq_stage(sm, cur ^ 1, Xhi, Xlo, Ehi, Elo, m0, c0, kk + 32, tid);
            vq_compute(sm, cur, acc, wm, wn, half, lrow);   // overlaps async DMA
            cur ^= 1;
        }
#else
        if (tid < 128) sm.en[tid] = enorm[c0 + tid];
        for (int kk = 0; kk < DDIM; kk += 32) {
            __syncthreads();
            vq_stage(sm, 0, Xhi, Xlo, Ehi, Elo, m0, c0, kk, tid);
            __syncthreads();
            vq_compute(sm, 0, acc, wm, wn, half, lrow);
        }
#endif

        // epilogue: s = ||e||^2 - 2*dot ; per-row argmin via 16-lane shuffle
        #pragma unroll
        for (int mi = 0; mi < 4; ++mi)
            #pragma unroll
            for (int ni = 0; ni < 2; ++ni) {
                const float en  = sm.en[wn * 32 + ni * 16 + lrow];
                const int  code = c0 + wn * 32 + ni * 16 + lrow;
                #pragma unroll
                for (int r = 0; r < 8; ++r) {
                    float s = __builtin_fmaf(-2.0f, acc[mi][ni][r], en);
                    int   c = code;
                    #pragma unroll
                    for (int off = 1; off < 16; off <<= 1) {
                        const float so = __shfl_xor(s, off, 32);
                        const int   co = __shfl_xor(c, off, 32);
                        if (so < s || (so == s && co < c)) { s = so; c = co; }
                    }
                    if (lrow == 0) {
                        const int rl = mi * 16 + half * 8 + r;
                        if (s < sm.wv[w][rl] ||
                            (s == sm.wv[w][rl] && c < sm.wi[w][rl])) {
                            sm.wv[w][rl] = s;
                            sm.wi[w][rl] = c;
                        }
                    }
                }
            }
    }

    __syncthreads();
    if (tid < 128) {
        const int wmg = tid >> 6;
        const int rl  = tid & 63;
        float bv = 3.0e38f;
        int   bc = 0x7fffffff;
        #pragma unroll
        for (int g = 0; g < 4; ++g) {
            const int w2 = g * 2 + wmg;
            const float v = sm.wv[w2][rl];
            const int   c = sm.wi[w2][rl];
            if (v < bv || (v == bv && c < bc)) { bv = v; bc = c; }
        }
        const unsigned int bits = __float_as_uint(bv);
        const unsigned int key  = bits ^ ((bits & 0x80000000u) ? 0xFFFFFFFFu : 0x80000000u);
        const unsigned long long packed = ((unsigned long long)key << 32) | (unsigned int)bc;
        atomicMin(&best[m0 + tid], packed);
    }
}

// ------- gather codes, write out + indices + loss partials (coalesced) ------
// grid = 512 blocks, one per (b,h); covers 32 points x 256 channels.
__global__ __launch_bounds__(256)
void vq_gather_out(const float* __restrict__ x, const float* __restrict__ emb,
                   const unsigned long long* __restrict__ best,
                   float* __restrict__ out, int* __restrict__ idx_out,
                   float* __restrict__ partial) {
    __shared__ int   codes[32];
    __shared__ float erows[32 * 257];          // [w][c], pitch 257
    __shared__ float red[256];
    const int bh  = blockIdx.x;                // b*32 + h
    const int tid = threadIdx.x;
    const int n0  = bh << 5;                   // first point of this block

    if (tid < 32) {
        const int code = (int)(unsigned int)(best[n0 + tid] & 0xFFFFFFFFull);
        codes[tid]     = code;
        idx_out[n0 + tid] = code;
    }
    __syncthreads();

    // stage the 32 selected codebook rows (coalesced 1KB loads per row)
    #pragma unroll 4
    for (int i = 0; i < 32; ++i) {
        erows[i * 257 + tid] = emb[(size_t)codes[i] * DDIM + tid];
    }
    __syncthreads();

    // per-element: lanes vary w -> coalesced x reads & out writes
    const int w  = tid & 31;
    const int c0 = tid >> 5;                   // 0..7
    const size_t xbase = (((size_t)bh >> 5) * DDIM) * 1024 +
                         (((size_t)bh & 31) << 5) + w;
    float sq = 0.0f;
    #pragma unroll
    for (int j = 0; j < 32; ++j) {
        const int c = c0 + j * 8;
        const size_t xo = xbase + (size_t)c * 1024;
        const float xv = x[xo];
        const float e  = erows[w * 257 + c];   // conflict-free: bank = (w+c)%64
        out[xo] = xv + (e - xv);               // straight-through estimator
        const float d = e - xv;
        sq += d * d;
    }

    red[tid] = sq;
    __syncthreads();
    for (int s = 128; s > 0; s >>= 1) {
        if (tid < s) red[tid] += red[tid + s];
        __syncthreads();
    }
    if (tid == 0) partial[bh] = red[0];
}

// ---------------- deterministic final loss reduction ------------------------
__global__ __launch_bounds__(256)
void vq_loss_reduce(const float* __restrict__ partial, float* __restrict__ loss) {
    const int t = threadIdx.x;
    float s = partial[t] + partial[t + 256];   // 512 partials
    __shared__ float red[256];
    red[t] = s;
    __syncthreads();
    for (int k = 128; k > 0; k >>= 1) {
        if (t < k) red[t] += red[t + k];
        __syncthreads();
    }
    if (t == 0) *loss = red[0] * (1.25f / 4194304.0f);  // (1+BETA)*mean
}

// ---------------------------------------------------------------------------
extern "C" void kernel_launch(void* const* d_in, const int* in_sizes, int n_in,
                              void* d_out, int out_size, void* d_ws, size_t ws_size,
                              hipStream_t stream) {
    const float* x   = (const float*)d_in[0];   // [16,256,32,32]
    const float* emb = (const float*)d_in[1];   // [8192,256]
    float* out = (float*)d_out;                 // 4194304 floats
    float* loss = out + 4194304;                // 1 float
    int*   idx_out = (int*)d_out + 4194305;     // 16384 int32

    char* ws = (char*)d_ws;
    __bf16* Xhi = (__bf16*)(ws + 0);            // 8 MB
    __bf16* Xlo = (__bf16*)(ws + 8388608);      // 8 MB
    __bf16* Ehi = (__bf16*)(ws + 16777216);     // 4 MB
    __bf16* Elo = (__bf16*)(ws + 20971520);     // 4 MB
    float*  enorm = (float*)(ws + 25165824);    // 32 KB
    unsigned long long* best = (unsigned long long*)(ws + 25198592); // 128 KB
    float*  partial = (float*)(ws + 25329664);  // 2 KB (512 block partials)

    hipLaunchKernelGGL(vq_init_best, dim3(64), dim3(256), 0, stream, best);
    hipLaunchKernelGGL(vq_prep_x, dim3(512), dim3(256), 0, stream, x, Xhi, Xlo);
    hipLaunchKernelGGL(vq_prep_emb, dim3(NCODE), dim3(256), 0, stream, emb, Ehi, Elo, enorm);
    hipLaunchKernelGGL(vq_gemm_argmin, dim3(NPTS / 128, NCODE / 1024), dim3(256), 0, stream,
                       Xhi, Xlo, Ehi, Elo, enorm, best);
    hipLaunchKernelGGL(vq_gather_out, dim3(512), dim3(256), 0, stream,
                       x, emb, best, out, idx_out, partial);
    hipLaunchKernelGGL(vq_loss_reduce, dim3(1), dim3(256), 0, stream, partial, loss);
}
